// _RelationalGATLayer_9852654977185
// MI455X (gfx1250) — matrix operand order repaired and checked by reading, not compiled, not run on hardware
//
#include <hip/hip_runtime.h>
#include <math.h>

#define DD 128
#define HH 8

#ifndef USE_TDM
#define USE_TDM 1   // stage W^T into LDS via tensor_load_to_lds (TDM)
#endif

typedef __attribute__((ext_vector_type(16))) _Float16 v16h;
typedef __attribute__((ext_vector_type(8)))  float    v8f;
typedef unsigned int u32;
typedef __attribute__((ext_vector_type(4))) u32 u32x4;
typedef __attribute__((ext_vector_type(8))) u32 u32x8;
typedef __attribute__((address_space(3))) _Float16 lds_half;

// ---- ds_swizzle xor-mode butterfly step (imm = (xor_mask<<10) | 0x1f) ----
template <int IMM>
__device__ __forceinline__ float swz_add(float v) {
  return v + __int_as_float(__builtin_amdgcn_ds_swizzle(__float_as_int(v), IMM));
}
// reduce over a 16-lane half (xor 8,4,2,1)
__device__ __forceinline__ float reduce16(float v) {
  v = swz_add<0x201F>(v);
  v = swz_add<0x101F>(v);
  v = swz_add<0x081F>(v);
  v = swz_add<0x041F>(v);
  return v;
}
// reduce over the full wave32 (xor 16,8,4,2,1)
__device__ __forceinline__ float reduce32(float v) {
  v = swz_add<0x401F>(v);
  return reduce16(v);
}

// ---- order-preserving float <-> uint for atomicMax-based segment max ----
__device__ __forceinline__ unsigned f2ord(float f) {
  unsigned u = __float_as_uint(f);
  return (u & 0x80000000u) ? ~u : (u | 0x80000000u);
}
__device__ __forceinline__ float ord2f(unsigned u) {
  return (u & 0x80000000u) ? __uint_as_float(u & 0x7fffffffu)
                           : __uint_as_float(~u);
}

// =====================================================================
// One-time: h -> f16 copy (shared by all three relations).
// =====================================================================
__global__ void conv_h(const float* __restrict__ h, _Float16* __restrict__ h16,
                       int total) {
  int t = blockIdx.x * blockDim.x + threadIdx.x;
  if (t < total) h16[t] = (_Float16)h[t];
}

// Per-relation: W (row-major [k][n]) -> transposed f16 w16[n][k].
__global__ void conv_w(const float* __restrict__ W, _Float16* __restrict__ w16) {
  int t = blockIdx.x * blockDim.x + threadIdx.x;
  if (t < DD * DD) {
    int k = t >> 7, n = t & 127;
    w16[n * DD + k] = (_Float16)W[t];
  }
}

// =====================================================================
// Per-relation init: m = -inf (ordered-uint), den = 0, gat accum = 0,
// and per-head edge-attention constants c[h] = sum_k We[h,k]*ae[h,k].
// =====================================================================
__global__ void rel_init(unsigned* __restrict__ m_ord, float* __restrict__ den,
                         float* __restrict__ gat, float* __restrict__ c_edge,
                         const float* __restrict__ We, const float* __restrict__ ae,
                         int Nn) {
  int t = blockIdx.x * blockDim.x + threadIdx.x;
  if (t < Nn * HH) { m_ord[t] = f2ord(-INFINITY); den[t] = 0.f; }
  if (t < Nn * DD) gat[t] = 0.f;
  if (t < HH) {
    float s = 0.f;
    if (We != nullptr) {
      #pragma unroll
      for (int k = 0; k < 16; ++k) s += We[t * 16 + k] * ae[t * 16 + k];
    }
    c_edge[t] = s;
  }
}

// =====================================================================
// WMMA GEMM: xp = h @ W (f16 in, f32 acc), fused attention logits.
// Block = 256 threads = 8 waves; tile = 64 rows x 128 cols.
// Wave w owns columns [16w,16w+16) == head w, 4 M-tiles -> 16 WMMAs.
// W^T (32 KB, raw f16) staged into LDS by the Tensor Data Mover.
// =====================================================================
__global__ void __launch_bounds__(256)
gat_gemm(const _Float16* __restrict__ h16, const _Float16* __restrict__ w16,
         const float* __restrict__ att_s, const float* __restrict__ att_d,
         float* __restrict__ xp, float* __restrict__ a_s, float* __restrict__ a_d,
         int Nn) {
  __shared__ __align__(16) _Float16 sW[DD * DD];  // W^T: sW[n][k]   (32 KB)
  __shared__ __align__(16) _Float16 sH[64 * DD];  // h tile: sH[r][k](16 KB)

  const int tid   = threadIdx.x;
  const int node0 = blockIdx.x * 64;

#if USE_TDM
  if (tid < 32) {
    // Build the Tensor DMA Descriptor (D#): 1-D tile of 8192 dwords
    // (32 KB) from w16 -> LDS sW.  group0: count=1 | lds_addr |
    // global_addr(57b) | type=2.  group1: data_size=4B, dims/strides.
    u32 ldsoff = (u32)(unsigned long long)(lds_half*)sW;
    unsigned long long ga = (unsigned long long)w16;
    u32x4 g0;
    g0[0] = 1u;                               // count = 1 valid descriptor
    g0[1] = ldsoff;                           // lds_addr (bytes)
    g0[2] = (u32)ga;                          // global_addr[31:0]
    g0[3] = (u32)(ga >> 32) | (2u << 30);     // global_addr[56:32], type=2
    u32x8 g1;
    g1[0] = (2u << 16);                       // data_size = 2 (4 bytes)
    g1[1] = (8192u << 16);                    // tensor_dim0[15:0] = 8192
    g1[2] = (1u << 16);                       // tensor_dim1 = 1
    g1[3] = (8192u << 16);                    // tile_dim0 = 8192
    g1[4] = 1u;                               // tile_dim1 = 1 (2-D tile)
    g1[5] = 8192u;                            // tensor_dim0_stride
    g1[6] = (8192u << 16);                    // tensor_dim1_stride[15:0]
    g1[7] = 0u;
    asm volatile("tensor_load_to_lds %0, %1" :: "s"(g0), "s"(g1) : "memory");
#if __has_builtin(__builtin_amdgcn_s_wait_tensorcnt)
    __builtin_amdgcn_s_wait_tensorcnt(0);
#else
    asm volatile("s_wait_tensorcnt 0x0" ::: "memory");
#endif
  }
#else
  for (int idx = tid; idx < (DD * DD) / 8; idx += 256)
    *(uint4*)&sW[idx * 8] = *(const uint4*)&w16[idx * 8];
#endif

  // h tile: 64 rows x 256 B, raw f16, 16-byte chunks.
  for (int idx = tid; idx < (64 * DD) / 8; idx += 256) {
    int r = idx >> 4, cc = idx & 15;
    int row = node0 + r; if (row >= Nn) row = Nn - 1;
    *(uint4*)&sH[r * DD + cc * 8] =
        *(const uint4*)&h16[(size_t)row * DD + cc * 8];
  }
  __syncthreads();

  const int wave = tid >> 5, lane = tid & 31;
  const int m = lane & 15, sel = lane >> 4;   // sel: which lane-half
  const int n0 = wave * 16;                   // column block == head id

  v8f c[4] = {{}, {}, {}, {}};
  #pragma unroll
  for (int kt = 0; kt < 4; ++kt) {            // K = 128 in 4 x 32 steps
    const int k0 = kt * 32;
    // B fragment (32x16): lane = column n; sel=0 -> K k0..k0+15, sel=1 -> +16
    v16h b;
    const _Float16* wr = &sW[(n0 + m) * DD];
    #pragma unroll
    for (int j = 0; j < 16; ++j) b[j] = wr[k0 + sel * 16 + j];
    #pragma unroll
    for (int mt = 0; mt < 4; ++mt) {
      // A fragment (16x32): sel=0 -> K 0..15, sel=1 -> 16..31
      v16h a;
      const _Float16* hr = &sH[(mt * 16 + m) * DD];
      #pragma unroll
      for (int i = 0; i < 8; ++i) {
        a[i]     = hr[k0 + sel * 16 + i];
        a[i + 8] = hr[k0 + sel * 16 + 8 + i];
      }
      c[mt] = __builtin_amdgcn_wmma_f32_16x16x32_f16(
          false, a, false, b, (short)0, c[mt], false, false);
    }
  }

  // Epilogue: store xp; reduce attention logits for head `wave`.
  const float asv = att_s[n0 + m];            // att_src[head=wave][elem=m]
  const float adv = att_d[n0 + m];
  #pragma unroll
  for (int mt = 0; mt < 4; ++mt) {
    #pragma unroll
    for (int r = 0; r < 8; ++r) {
      const int row = node0 + mt * 16 + sel * 8 + r;  // C: M=8*sel+r, N=m
      if (row < Nn) xp[row * DD + n0 + m] = c[mt][r];
      float vs = reduce16(c[mt][r] * asv);
      float vd = reduce16(c[mt][r] * adv);
      if (m == 0 && row < Nn) {
        a_s[row * HH + wave] = vs;
        a_d[row * HH + wave] = vd;
      }
    }
  }
}

// ---- alpha recompute: leaky_relu(a_s[src]+a_d[dst]+w_e*c_h, 0.2) ----
__device__ __forceinline__ float edge_alpha(const float* a_s, const float* a_d,
                                            const float* ew, const float* c_edge,
                                            int s, int d, int e, int hh) {
  float al = a_s[s * HH + hh] + a_d[d * HH + hh];
  if (ew != nullptr) al += ew[e] * c_edge[hh];
  return al > 0.f ? al : 0.2f * al;
}

// Pass 1: segment max over destinations (ordered-uint atomicMax).
__global__ void edge_max(const int* __restrict__ src, const int* __restrict__ dst,
                         const float* __restrict__ ew,
                         const float* __restrict__ a_s, const float* __restrict__ a_d,
                         const float* __restrict__ c_edge,
                         unsigned* __restrict__ m_ord, int Ee) {
  int t = blockIdx.x * blockDim.x + threadIdx.x;
  if (t >= Ee * HH) return;
  int e = t >> 3, hh = t & 7;
  int s = src[e], d = dst[e];
  float al = edge_alpha(a_s, a_d, ew, c_edge, s, d, e, hh);
  atomicMax(&m_ord[d * HH + hh], f2ord(al));
}

// Pass 2: segment sum of exp(alpha - max).
__global__ void edge_den(const int* __restrict__ src, const int* __restrict__ dst,
                         const float* __restrict__ ew,
                         const float* __restrict__ a_s, const float* __restrict__ a_d,
                         const float* __restrict__ c_edge,
                         const unsigned* __restrict__ m_ord,
                         float* __restrict__ den, int Ee) {
  int t = blockIdx.x * blockDim.x + threadIdx.x;
  if (t >= Ee * HH) return;
  int e = t >> 3, hh = t & 7;
  int s = src[e], d = dst[e];
  float al = edge_alpha(a_s, a_d, ew, c_edge, s, d, e, hh);
  float ex = __expf(al - ord2f(m_ord[d * HH + hh]));
  atomicAdd(&den[d * HH + hh], ex);
}

// Pass 3: messages. One wave per edge; lane l handles 4 features (head l>>2).
__global__ void edge_msg(const int* __restrict__ src, const int* __restrict__ dst,
                         const float* __restrict__ ew,
                         const float* __restrict__ a_s, const float* __restrict__ a_d,
                         const float* __restrict__ c_edge,
                         const unsigned* __restrict__ m_ord,
                         const float* __restrict__ den,
                         const float* __restrict__ xp, float* __restrict__ gat,
                         int Ee) {
  int t = blockIdx.x * blockDim.x + threadIdx.x;
  if (t >= Ee * 32) return;
  int e = t >> 5, l = t & 31;
  int s = src[e], d = dst[e];
  int hh = l >> 2;
  float al   = edge_alpha(a_s, a_d, ew, c_edge, s, d, e, hh);
  float ex   = __expf(al - ord2f(m_ord[d * HH + hh]));
  float coef = ex / (den[d * HH + hh] + 1e-16f);
  const float4 v = *(const float4*)&xp[s * DD + l * 4];
  float* o = &gat[d * DD + l * 4];
  atomicAdd(o + 0, v.x * coef);
  atomicAdd(o + 1, v.y * coef);
  atomicAdd(o + 2, v.z * coef);
  atomicAdd(o + 3, v.w * coef);
}

// =====================================================================
// Finalize one relation: out (+)= softmax(logits)[which] * elu(ln(gat+b)).
// Relation 0 adds the residual h and overwrites out (replay-safe).
// One wave per node; 4 features per lane; swizzle reductions for LN.
// =====================================================================
__global__ void __launch_bounds__(256)
rel_finalize(const float* __restrict__ gat, const float* __restrict__ bias,
             const float* __restrict__ g, const float* __restrict__ bvec,
             const float* __restrict__ hsrc, const float* __restrict__ logits,
             int which, int first, float* __restrict__ out, int Nn) {
  int gw   = (blockIdx.x * blockDim.x + threadIdx.x) >> 5;
  int lane = threadIdx.x & 31;
  if (gw >= Nn) return;

  float l0 = logits[0], l1 = logits[1], l2 = logits[2];
  float mx = fmaxf(l0, fmaxf(l1, l2));
  float e0 = __expf(l0 - mx), e1 = __expf(l1 - mx), e2 = __expf(l2 - mx);
  float sel3 = (which == 0) ? e0 : (which == 1) ? e1 : e2;
  float aw = sel3 / (e0 + e1 + e2);

  float y[4];
  float sum = 0.f;
  #pragma unroll
  for (int i = 0; i < 4; ++i) {
    int j = lane * 4 + i;
    y[i] = gat[gw * DD + j] + bias[j];
    sum += y[i];
  }
  sum = reduce32(sum);
  float mu = sum * (1.f / 128.f);

  float var = 0.f;
  #pragma unroll
  for (int i = 0; i < 4; ++i) { float dd = y[i] - mu; var += dd * dd; }
  var = reduce32(var) * (1.f / 128.f);
  float inv = rsqrtf(var + 1e-5f);

  #pragma unroll
  for (int i = 0; i < 4; ++i) {
    int j = lane * 4 + i;
    float v = g[j] * (y[i] - mu) * inv + bvec[j];
    v = v > 0.f ? v : (__expf(v) - 1.f);          // elu, alpha=1
    float base = first ? hsrc[gw * DD + j] : out[gw * DD + j];
    out[gw * DD + j] = base + aw * v;
  }
}

// =====================================================================
extern "C" void kernel_launch(void* const* d_in, const int* in_sizes, int n_in,
                              void* d_out, int out_size, void* d_ws, size_t ws_size,
                              hipStream_t stream) {
  (void)n_in; (void)out_size; (void)ws_size;
  const int Nn = in_sizes[0] / DD;   // 50000
  const int Ee = in_sizes[3];        // 500000

  const float* h      = (const float*)d_in[0];
  const float* logits = (const float*)d_in[28];
  float* out = (float*)d_out;

  // workspace carve-out (reused across the 3 relations): ~71 MB
  char* ws = (char*)d_ws;
  size_t off = 0;
  auto take = [&](size_t bytes) -> char* {
    char* p = ws + off;
    off = (off + bytes + 255) & ~(size_t)255;
    return p;
  };
  float*    xp     = (float*)   take((size_t)Nn * DD * 4);
  float*    a_s    = (float*)   take((size_t)Nn * HH * 4);
  float*    a_d    = (float*)   take((size_t)Nn * HH * 4);
  unsigned* m_ord  = (unsigned*)take((size_t)Nn * HH * 4);
  float*    den    = (float*)   take((size_t)Nn * HH * 4);
  float*    gat    = (float*)   take((size_t)Nn * DD * 4);
  _Float16* h16    = (_Float16*)take((size_t)Nn * DD * 2);
  _Float16* w16    = (_Float16*)take((size_t)DD * DD * 2);
  float*    c_edge = (float*)   take(64);

  struct Rel {
    const int* idx; const float* ew;
    const float* W; const float* as_; const float* ad_; const float* b;
    const float* ae; const float* We; const float* g; const float* bv;
  };
  Rel rels[3] = {
    { (const int*)d_in[1], nullptr,
      (const float*)d_in[6], (const float*)d_in[7], (const float*)d_in[8], (const float*)d_in[9],
      nullptr, nullptr, (const float*)d_in[22], (const float*)d_in[23] },
    { (const int*)d_in[2], (const float*)d_in[3],
      (const float*)d_in[10], (const float*)d_in[11], (const float*)d_in[12], (const float*)d_in[13],
      (const float*)d_in[18], (const float*)d_in[19], (const float*)d_in[24], (const float*)d_in[25] },
    { (const int*)d_in[4], (const float*)d_in[5],
      (const float*)d_in[14], (const float*)d_in[15], (const float*)d_in[16], (const float*)d_in[17],
      (const float*)d_in[20], (const float*)d_in[21], (const float*)d_in[26], (const float*)d_in[27] },
  };

  conv_h<<<(Nn * DD + 255) / 256, 256, 0, stream>>>(h, h16, Nn * DD);

  for (int r = 0; r < 3; ++r) {
    const Rel& R = rels[r];
    const int* src = R.idx;
    const int* dst = R.idx + Ee;

    rel_init<<<(Nn * DD + 255) / 256, 256, 0, stream>>>(m_ord, den, gat, c_edge,
                                                        R.We, R.ae, Nn);
    conv_w<<<(DD * DD + 255) / 256, 256, 0, stream>>>(R.W, w16);
    gat_gemm<<<(Nn + 63) / 64, 256, 0, stream>>>(h16, w16, R.as_, R.ad_,
                                                 xp, a_s, a_d, Nn);
    int tEH = Ee * HH;
    edge_max<<<(tEH + 255) / 256, 256, 0, stream>>>(src, dst, R.ew, a_s, a_d,
                                                    c_edge, m_ord, Ee);
    edge_den<<<(tEH + 255) / 256, 256, 0, stream>>>(src, dst, R.ew, a_s, a_d,
                                                    c_edge, m_ord, den, Ee);
    long tEM = (long)Ee * 32;
    edge_msg<<<(int)((tEM + 255) / 256), 256, 0, stream>>>(src, dst, R.ew, a_s, a_d,
                                                           c_edge, m_ord, den,
                                                           xp, gat, Ee);
    rel_finalize<<<(Nn * 32 + 255) / 256, 256, 0, stream>>>(gat, R.b, R.g, R.bv,
                                                            h, logits, r, r == 0,
                                                            out, Nn);
  }
}
